// QuantumAttention_39960375722235
// MI455X (gfx1250) — matrix operand order repaired and checked by reading, not compiled, not run on hardware
//
#include <hip/hip_runtime.h>
#include <hip/hip_bf16.h>

#define DEVINL __device__ __forceinline__

typedef __attribute__((ext_vector_type(16))) __bf16 v16bf;
typedef __attribute__((ext_vector_type(8)))  float  v8f;

constexpr int B_  = 2;
constexpr int S_  = 2048;
constexpr int D_  = 1024;
constexpr int H_  = 16;
constexpr int DK_ = 64;
constexpr int M_  = B_ * S_;   // 4096 activation rows

// ---------------------------------------------------------------------------
// CDNA5 16-bit WMMA fragment loaders (ISA 7.12.2 layouts, wave32)
// A 16x32 (MxK): lane m (0..15)/m+16 holds row m; elems 0..7 = K[8*lh .. +7],
//                elems 8..15 = K[16+8*lh .. +7]  (lh = lane>=16)
// B 32x16 (KxN): lane n holds column n; elems 0..7 = K[8*lh .. +7], 8..15 = K[16+8*lh .. +7]
// C/D 16x16:     VGPR v, lanes 0..15 -> row v, lanes 16..31 -> row v+8; col = lane&15
// ---------------------------------------------------------------------------
DEVINL v16bf ld_afrag(const __bf16* rowPtr, int k0, int lhalf) {
  union { int4 q[2]; v16bf v; } u;
  u.q[0] = *(const int4*)(rowPtr + k0 + 8 * lhalf);
  u.q[1] = *(const int4*)(rowPtr + k0 + 16 + 8 * lhalf);
  return u.v;
}
DEVINL v16bf ld_bfrag(const __bf16* colPtr, int k0, int lhalf) {
  union { int4 q[2]; v16bf v; } u;
  u.q[0] = *(const int4*)(colPtr + k0 + 8 * lhalf);
  u.q[1] = *(const int4*)(colPtr + k0 + 16 + 8 * lhalf);
  return u.v;
}
DEVINL v8f wmma_bf16(v16bf a, v16bf b, v8f c) {
  return __builtin_amdgcn_wmma_f32_16x16x32_bf16(false, a, false, b,
                                                 (short)0, c, false, false);
}

// Async copy of one 64B chunk per lane: LDS[lds_off + i*16] = base[gl_off + i*16]
// GVS mode: mem = SADDR + VADDR32 + IOFFSET; IOFFSET advances LDS and global
// addresses in lockstep (ISA 10.x async pseudocode). Tracked by ASYNCcnt.
DEVINL void async_copy64(unsigned lds_off, unsigned gl_off, const void* base) {
  asm volatile(
      "global_load_async_to_lds_b128 %0, %1, %2\n\t"
      "global_load_async_to_lds_b128 %0, %1, %2 offset:16\n\t"
      "global_load_async_to_lds_b128 %0, %1, %2 offset:32\n\t"
      "global_load_async_to_lds_b128 %0, %1, %2 offset:48"
      :: "v"(lds_off), "v"(gl_off), "s"(base) : "memory");
}
DEVINL void wait_async0() {
  asm volatile("s_wait_asynccnt 0x0" ::: "memory");
}

enum { MODE_QK = 0, MODE_VT = 1, MODE_OUT = 2 };

// ---------------------------------------------------------------------------
// C = A[M,K] @ W[N,K]^T + bias, 128x128 block tile, 8 waves (2x4), K-step 32.
// f32 -> bf16 conversion happens while staging tiles to LDS.
// ---------------------------------------------------------------------------
template <int MODE>
__global__ __launch_bounds__(256)
void gemm_wmma_kernel(const float* __restrict__ A,
                      const float* __restrict__ W,
                      const float* __restrict__ bias,
                      void* __restrict__ out,
                      int Kdim, int Ncols)
{
  __shared__ __bf16 As[128][40];   // 80B row stride: conflict-free, 16B aligned
  __shared__ __bf16 Ws[128][40];

  const int tid   = threadIdx.x;
  const int lane  = tid & 31;
  const int wave  = tid >> 5;
  const int lhalf = lane >> 4;
  const int l16   = lane & 15;
  const int wm    = wave & 1;          // 2 waves along M (64 rows each)
  const int wn    = wave >> 1;         // 4 waves along N (32 cols each)
  const int rowBase = blockIdx.x * 128;
  const int colBase = blockIdx.y * 128;

  v8f acc[4][2] = {};

  const int ldr = tid >> 1;            // 0..127: tile row loaded by this thread
  const int ldc = (tid & 1) * 16;      // 0 / 16: 16-element half-row

  for (int k0 = 0; k0 < Kdim; k0 += 32) {
    {
      const float4* s4 = (const float4*)(A + (size_t)(rowBase + ldr) * Kdim + k0 + ldc);
      union { __bf16 b[16]; int4 q[2]; } cv;
      #pragma unroll
      for (int p = 0; p < 4; ++p) {
        float4 f = s4[p];
        cv.b[p*4+0] = (__bf16)f.x; cv.b[p*4+1] = (__bf16)f.y;
        cv.b[p*4+2] = (__bf16)f.z; cv.b[p*4+3] = (__bf16)f.w;
      }
      *(int4*)&As[ldr][ldc]     = cv.q[0];
      *(int4*)&As[ldr][ldc + 8] = cv.q[1];
    }
    {
      const float4* s4 = (const float4*)(W + (size_t)(colBase + ldr) * Kdim + k0 + ldc);
      union { __bf16 b[16]; int4 q[2]; } cv;
      #pragma unroll
      for (int p = 0; p < 4; ++p) {
        float4 f = s4[p];
        cv.b[p*4+0] = (__bf16)f.x; cv.b[p*4+1] = (__bf16)f.y;
        cv.b[p*4+2] = (__bf16)f.z; cv.b[p*4+3] = (__bf16)f.w;
      }
      *(int4*)&Ws[ldr][ldc]     = cv.q[0];
      *(int4*)&Ws[ldr][ldc + 8] = cv.q[1];
    }
    __syncthreads();

    v16bf afrag[4], bfrag[2];
    #pragma unroll
    for (int i = 0; i < 4; ++i)
      afrag[i] = ld_afrag(&As[wm * 64 + i * 16 + l16][0], 0, lhalf);
    #pragma unroll
    for (int j = 0; j < 2; ++j)
      bfrag[j] = ld_bfrag(&Ws[wn * 32 + j * 16 + l16][0], 0, lhalf);
    #pragma unroll
    for (int i = 0; i < 4; ++i)
      #pragma unroll
      for (int j = 0; j < 2; ++j)
        acc[i][j] = wmma_bf16(afrag[i], bfrag[j], acc[i][j]);
    __syncthreads();
  }

  // ---- epilogue -----------------------------------------------------------
  #pragma unroll
  for (int i = 0; i < 4; ++i) {
    #pragma unroll
    for (int j = 0; j < 2; ++j) {
      const int n  = colBase + wn * 32 + j * 16 + l16;
      const float bn = bias[n];
      const int m0 = rowBase + wm * 64 + i * 16 + 8 * lhalf;  // rows m0..m0+7

      if (MODE == MODE_OUT) {
        float* O = (float*)out;
        #pragma unroll
        for (int v = 0; v < 8; ++v)
          O[(size_t)(m0 + v) * Ncols + n] = acc[i][j][v] + bn;
      } else if (MODE == MODE_QK) {
        __bf16* O = (__bf16*)out;               // [B,H,S,DK]
        const int h = n >> 6, d = n & 63;
        const int b = m0 / S_, s0 = m0 % S_;
        #pragma unroll
        for (int v = 0; v < 8; ++v)
          O[((size_t)(b * H_ + h) * S_ + (s0 + v)) * DK_ + d] =
              (__bf16)(acc[i][j][v] + bn);
      } else {                                   // MODE_VT: [B,H,DK,S]
        __bf16* O = (__bf16*)out;
        const int h = n >> 6, d = n & 63;
        const int b = m0 / S_, s0 = m0 % S_;
        union { __bf16 hh[8]; int4 q; } pk;
        #pragma unroll
        for (int v = 0; v < 8; ++v) pk.hh[v] = (__bf16)(acc[i][j][v] + bn);
        *(int4*)&O[((size_t)(b * H_ + h) * DK_ + d) * S_ + s0] = pk.q;
      }
    }
  }
}

// ---------------------------------------------------------------------------
// Flash attention with the quantum score transform.
// Block: 128 threads = 4 waves, Br=64 query rows (16 per wave), Bc=64 keys.
// K/V^T tiles are staged with CDNA5 async-to-LDS copies (ASYNCcnt).
// ---------------------------------------------------------------------------
__global__ __launch_bounds__(128)
void quantum_attn_kernel(const __bf16* __restrict__ Qb,   // [B,H,S,DK]
                         const __bf16* __restrict__ Kb,   // [B,H,S,DK]
                         const __bf16* __restrict__ Vt,   // [B,H,DK,S]
                         const float* __restrict__ phases,
                         const float* __restrict__ imag01,
                         float* __restrict__ ctx)         // [B,S,D] f32
{
  __shared__ __bf16 Ks[64][72];        // [key][d], 144B stride (conflict-free)
  __shared__ __bf16 Vs[64][72];        // [d][key]
  __shared__ __bf16 Ps[4][16][72];     // wave-private P staging

  const int tid   = threadIdx.x;
  const int lane  = tid & 31;
  const int wave  = tid >> 5;
  const int lhalf = lane >> 4;
  const int l16   = lane & 15;
  const int bh    = blockIdx.y;        // b*H + h
  const int h     = bh & (H_ - 1);
  const int b     = bh >> 4;
  const int q0    = blockIdx.x * 64;

  const __bf16* Qh = Qb + (size_t)bh * S_ * DK_;
  const __bf16* Kh = Kb + (size_t)bh * S_ * DK_;
  const __bf16* Vh = Vt + (size_t)bh * DK_ * S_;

  const float theta = phases[h];
  const float ic    = imag01[h];       // 0.1 * imag[h]
  const float scale = 0.125f;          // 1/sqrt(DK)

  // persistent Q fragments for this wave's 16 rows
  const int qrow = q0 + wave * 16 + l16;
  v16bf qf[2];
  qf[0] = ld_afrag(Qh + (size_t)qrow * DK_, 0,  lhalf);
  qf[1] = ld_afrag(Qh + (size_t)qrow * DK_, 32, lhalf);

  float rm[8], rl[8];
  v8f o[4] = {};
  #pragma unroll
  for (int v = 0; v < 8; ++v) { rm[v] = -3.0e38f; rl[v] = 0.f; }

  const int ldr = tid >> 1;            // 0..63: tile row staged by this thread
  const int ldc = (tid & 1) * 32;      // 0 / 32: 32-element (64B) half-row

  // LDS byte offsets of this thread's staging chunks (low 32 bits of a
  // generic pointer to __shared__ data are the LDS offset)
  const unsigned ldsK = (unsigned)(size_t)&Ks[ldr][ldc];
  const unsigned ldsV = (unsigned)(size_t)&Vs[ldr][ldc];
  const unsigned glV  = (unsigned)(((size_t)ldr * S_ + ldc) * sizeof(__bf16));

  for (int kk = 0; kk < S_; kk += 64) {
    // async-stage K tile [64 keys][64 d] and V^T tile [64 d][64 keys]
    const unsigned glK = (unsigned)(((size_t)(kk + ldr) * DK_ + ldc) * sizeof(__bf16));
    async_copy64(ldsK, glK, Kh);
    async_copy64(ldsV, glV + (unsigned)(kk * sizeof(__bf16)), Vh);
    if (kk + 64 < S_)
      __builtin_prefetch(Kh + (size_t)(kk + 64 + ldr) * DK_ + ldc, 0, 1);
    wait_async0();
    __syncthreads();

    // scores: 16x64 per wave
    v8f sc[4] = {};
    #pragma unroll
    for (int t = 0; t < 4; ++t)
      #pragma unroll
      for (int c = 0; c < 2; ++c) {
        v16bf kf = ld_bfrag(&Ks[t * 16 + l16][0], c * 32, lhalf);
        sc[t] = wmma_bf16(qf[c], kf, sc[t]);
      }

    // quantum transform: qe = s*cos(theta + 0.1 s) + 0.1*imag
    #pragma unroll
    for (int t = 0; t < 4; ++t)
      #pragma unroll
      for (int v = 0; v < 8; ++v) {
        float s = sc[t][v] * scale;
        sc[t][v] = s * __cosf(theta + 0.1f * s) + ic;
      }

    // online softmax (rows live across 16 lanes of one half + 4 n-tiles)
    #pragma unroll
    for (int v = 0; v < 8; ++v) {
      float mx = fmaxf(fmaxf(sc[0][v], sc[1][v]), fmaxf(sc[2][v], sc[3][v]));
      #pragma unroll
      for (int off = 1; off < 16; off <<= 1)
        mx = fmaxf(mx, __shfl_xor(mx, off, 32));
      const float mnew = fmaxf(rm[v], mx);
      const float corr = __expf(rm[v] - mnew);
      rm[v] = mnew;
      float rs = 0.f;
      #pragma unroll
      for (int t = 0; t < 4; ++t) {
        float p = __expf(sc[t][v] - mnew);
        sc[t][v] = p;
        rs += p;
      }
      #pragma unroll
      for (int off = 1; off < 16; off <<= 1)
        rs += __shfl_xor(rs, off, 32);
      rl[v] = rl[v] * corr + rs;
      #pragma unroll
      for (int t = 0; t < 4; ++t) o[t][v] *= corr;
    }

    // P (C-layout) -> wave-private LDS -> A-layout fragments
    #pragma unroll
    for (int t = 0; t < 4; ++t)
      #pragma unroll
      for (int v = 0; v < 8; ++v)
        Ps[wave][v + 8 * lhalf][t * 16 + l16] = (__bf16)sc[t][v];
    asm volatile("s_wait_dscnt 0" ::: "memory");

    // O += P @ V
    #pragma unroll
    for (int c = 0; c < 2; ++c) {
      v16bf pf = ld_afrag(&Ps[wave][l16][0], c * 32, lhalf);
      #pragma unroll
      for (int t = 0; t < 4; ++t) {
        v16bf vf = ld_bfrag(&Vs[t * 16 + l16][0], c * 32, lhalf);
        o[t] = wmma_bf16(pf, vf, o[t]);
      }
    }
    __syncthreads();
  }

  // normalize, write f32 context [B,S,D] for the output GEMM
  #pragma unroll
  for (int v = 0; v < 8; ++v) {
    const float inv  = 1.0f / rl[v];
    const int   srow = q0 + wave * 16 + v + 8 * lhalf;
    float* dst = ctx + (size_t)(b * S_ + srow) * D_ + h * DK_;
    #pragma unroll
    for (int t = 0; t < 4; ++t)
      dst[t * 16 + l16] = o[t][v] * inv;
  }
}

__global__ void precompute_imag_kernel(const float* __restrict__ phases,
                                       const float* __restrict__ wr,
                                       const float* __restrict__ wi,
                                       float* __restrict__ imag01)
{
  const int h = threadIdx.x;
  if (h < H_) {
    float s = 0.f;
    for (int d = 0; d < DK_; ++d) s += wr[h * DK_ + d] * wi[h * DK_ + d];
    imag01[h] = 0.1f * s * __sinf(phases[h]);
  }
}

// ---------------------------------------------------------------------------
extern "C" void kernel_launch(void* const* d_in, const int* in_sizes, int n_in,
                              void* d_out, int out_size, void* d_ws, size_t ws_size,
                              hipStream_t stream)
{
  (void)in_sizes; (void)n_in; (void)out_size; (void)ws_size;

  const float* query  = (const float*)d_in[0];
  const float* key_   = (const float*)d_in[1];
  const float* value  = (const float*)d_in[2];
  const float* Wq     = (const float*)d_in[3];
  const float* bq     = (const float*)d_in[4];
  const float* Wk     = (const float*)d_in[5];
  const float* bk     = (const float*)d_in[6];
  const float* Wv     = (const float*)d_in[7];
  const float* bv     = (const float*)d_in[8];
  const float* Wo     = (const float*)d_in[9];
  const float* bo     = (const float*)d_in[10];
  const float* phases = (const float*)d_in[11];
  const float* w_real = (const float*)d_in[12];
  const float* w_imag = (const float*)d_in[13];

  char* ws = (char*)d_ws;
  const size_t actBytes = (size_t)B_ * H_ * S_ * DK_ * sizeof(__bf16);  // 8 MiB
  __bf16* Qb  = (__bf16*)(ws);
  __bf16* Kb  = (__bf16*)(ws + actBytes);
  __bf16* Vtb = (__bf16*)(ws + 2 * actBytes);
  float*  ctx = (float*)(ws + 3 * actBytes);                      // 16 MiB f32
  float*  im  = (float*)(ws + 3 * actBytes + (size_t)M_ * D_ * sizeof(float));

  precompute_imag_kernel<<<1, 64, 0, stream>>>(phases, w_real, w_imag, im);

  dim3 gGrid(M_ / 128, D_ / 128);   // 32 x 8 blocks
  gemm_wmma_kernel<MODE_QK><<<gGrid, 256, 0, stream>>>(query, Wq, bq, Qb,  D_, D_);
  gemm_wmma_kernel<MODE_QK><<<gGrid, 256, 0, stream>>>(key_,  Wk, bk, Kb,  D_, D_);
  gemm_wmma_kernel<MODE_VT><<<gGrid, 256, 0, stream>>>(value, Wv, bv, Vtb, D_, D_);

  quantum_attn_kernel<<<dim3(S_ / 64, B_ * H_), 128, 0, stream>>>(
      Qb, Kb, Vtb, phases, im, ctx);

  gemm_wmma_kernel<MODE_OUT><<<gGrid, 256, 0, stream>>>(ctx, Wo, bo,
                                                        (float*)d_out, D_, D_);
}
